// RGAT_GCN_concat_22333829939347
// MI455X (gfx1250) — compile-verified, matrix operand order checked by reading, not run on hardware
//
#include <hip/hip_runtime.h>
#include <hip/hip_bf16.h>

// Problem constants (fixed by the reference)
#define N_NODES 50000
#define N_EDGES 800000
#define R_REL   3
#define DIM_IN  128
#define HD      64          // HID/2
#define NHEAD   4
#define OUT_DIM 64
#define SLOPE_ATTN 0.2f
#define SLOPE_ACT  0.01f

typedef __attribute__((ext_vector_type(16))) __bf16 v16bf;
typedef __attribute__((ext_vector_type(8)))  float  v8f;

// ---------------------------------------------------------------------------
// WMMA fragment loaders (ISA 7.12.2 layouts, bf16 16x16x32, f32 accum)
// A: 16x32 row-major f32 source; lane l<16 -> row l, k0=kb; lane>=16 -> k0=kb+8
//    vector elems [0..7] = K k0..k0+7, [8..15] = K k0+16..k0+23
// k0 is a multiple of 8 -> 32B aligned -> float4 loads are legal.
// ---------------------------------------------------------------------------
__device__ __forceinline__ v16bf load_a_frag(const float* __restrict__ A, int ldA,
                                             int row, int k0) {
    const float* p = A + (size_t)row * ldA + k0;
    const float4 a0 = *(const float4*)(p + 0);
    const float4 a1 = *(const float4*)(p + 4);
    const float4 a2 = *(const float4*)(p + 16);
    const float4 a3 = *(const float4*)(p + 20);
    v16bf a;
    a[0]  = (__bf16)a0.x; a[1]  = (__bf16)a0.y; a[2]  = (__bf16)a0.z; a[3]  = (__bf16)a0.w;
    a[4]  = (__bf16)a1.x; a[5]  = (__bf16)a1.y; a[6]  = (__bf16)a1.z; a[7]  = (__bf16)a1.w;
    a[8]  = (__bf16)a2.x; a[9]  = (__bf16)a2.y; a[10] = (__bf16)a2.z; a[11] = (__bf16)a2.w;
    a[12] = (__bf16)a3.x; a[13] = (__bf16)a3.y; a[14] = (__bf16)a3.z; a[15] = (__bf16)a3.w;
    return a;
}

// B: 32x16, source row-major KxldB f32; lane l<16 -> col l, same K chunking.
// Strided along K but coalesced across lanes (consecutive lanes -> consecutive cols).
__device__ __forceinline__ v16bf load_b_frag(const float* __restrict__ W, int ldB,
                                             int col, int k0) {
    v16bf b;
    const float* p = W + (size_t)k0 * ldB + col;
#pragma unroll
    for (int j = 0; j < 8; ++j) {
        b[j]     = (__bf16)p[j * ldB];
        b[8 + j] = (__bf16)p[(16 + j) * ldB];
    }
    return b;
}

// ---------------------------------------------------------------------------
// GEMM: feat[r] = A(NxK) @ Wall[r](Kx64), optional per-row scale (GCN "ns").
// block = (32 lanes, 4 waves = 4 col tiles), grid = (N/16, 1, R)
// K and USE_RS are compile-time: full unroll, no epilogue branches.
// ---------------------------------------------------------------------------
template <int K, bool USE_RS>
__global__ void gemm_feat_kernel(const float* __restrict__ A,
                                 const float* __restrict__ Wall,
                                 float* __restrict__ Out,
                                 const float* __restrict__ rs_all) {
    const int lane  = threadIdx.x;
    const int cTile = threadIdx.y;
    const int mTile = blockIdx.x;
    const int r     = blockIdx.z;

    const float* W = Wall + (size_t)r * K * HD;
    float*       C = Out  + (size_t)r * N_NODES * HD;

    const int half = lane >> 4;
    const int l    = lane & 15;
    const int rowA = mTile * 16 + l;
    const int colB = cTile * 16 + l;

    v8f c = {0.f, 0.f, 0.f, 0.f, 0.f, 0.f, 0.f, 0.f};
#pragma unroll
    for (int kb = 0; kb < K; kb += 32) {
        v16bf a = load_a_frag(A, K,  rowA, kb + half * 8);
        v16bf b = load_b_frag(W, HD, colB, kb + half * 8);
        c = __builtin_amdgcn_wmma_f32_16x16x32_bf16(false, a, false, b,
                                                    (short)0, c, false, false);
    }
    const int rowBase = mTile * 16 + half * 8;
    float s[8];
    if (USE_RS) {
        const float* rs = rs_all + (size_t)r * N_NODES + rowBase;
#pragma unroll
        for (int i = 0; i < 8; ++i) s[i] = rs[i];
    }
#pragma unroll
    for (int i = 0; i < 8; ++i) {
        float v = c[i];
        if (USE_RS) v *= s[i];
        C[(size_t)(rowBase + i) * HD + (cTile * 16 + l)] = v;
    }
}

// Final head: [h_gat | h_gcn](Nx128) @ Wlin(128x64) + blin, split-K (no concat)
__global__ void gemm_concat_kernel(const float* __restrict__ A0,
                                   const float* __restrict__ A1,
                                   const float* __restrict__ Wlin,
                                   const float* __restrict__ blin,
                                   float* __restrict__ Out) {
    const int lane  = threadIdx.x;
    const int cTile = threadIdx.y;
    const int mTile = blockIdx.x;
    const int half  = lane >> 4;
    const int l     = lane & 15;
    const int rowA  = mTile * 16 + l;
    const int colB  = cTile * 16 + l;

    v8f c = {0.f, 0.f, 0.f, 0.f, 0.f, 0.f, 0.f, 0.f};
#pragma unroll
    for (int kb = 0; kb < 128; kb += 32) {
        const float* A = (kb < 64) ? A0 : A1;      // each 32-chunk stays in one half
        v16bf a = load_a_frag(A, HD, rowA, (kb & 63) + half * 8);
        v16bf b = load_b_frag(Wlin, OUT_DIM, colB, kb + half * 8);
        c = __builtin_amdgcn_wmma_f32_16x16x32_bf16(false, a, false, b,
                                                    (short)0, c, false, false);
    }
    float bias = blin[cTile * 16 + l];
    const int rowBase = mTile * 16 + half * 8;
#pragma unroll
    for (int i = 0; i < 8; ++i) {
        Out[(size_t)(rowBase + i) * OUT_DIM + (cTile * 16 + l)] = c[i] + bias;
    }
}

// ---------------------------------------------------------------------------
// Elementwise / scatter kernels
// ---------------------------------------------------------------------------
__global__ void fill_kernel(float* __restrict__ p, long n, float v) {
    long i = (long)blockIdx.x * blockDim.x + threadIdx.x;
    if (i < n) p[i] = v;
}

// el[r,n,h] = sum_d feat[r,n,h,d]*al[r,h,d]; likewise er. float4 stores.
__global__ void attn_dots_kernel(const float* __restrict__ feat,
                                 const float* __restrict__ al_all,
                                 const float* __restrict__ ar_all,
                                 float* __restrict__ el, float* __restrict__ er) {
    long idx = (long)blockIdx.x * blockDim.x + threadIdx.x;  // over R*N
    if (idx >= (long)R_REL * N_NODES) return;
    int r = (int)(idx / N_NODES);
    const float* f  = feat   + idx * 64;
    const float* al = al_all + (size_t)r * 64;
    const float* ar = ar_all + (size_t)r * 64;
    float4 elv, erv;
    float* elp = &elv.x;
    float* erp = &erv.x;
#pragma unroll
    for (int h = 0; h < NHEAD; ++h) {
        float sl = 0.f, sr = 0.f;
#pragma unroll
        for (int d4 = 0; d4 < 4; ++d4) {
            const float4 fv  = *(const float4*)(f  + h * 16 + d4 * 4);
            const float4 alv = *(const float4*)(al + h * 16 + d4 * 4);
            const float4 arv = *(const float4*)(ar + h * 16 + d4 * 4);
            sl += fv.x * alv.x + fv.y * alv.y + fv.z * alv.z + fv.w * alv.w;
            sr += fv.x * arv.x + fv.y * arv.y + fv.z * arv.z + fv.w * arv.w;
        }
        elp[h] = sl;
        erp[h] = sr;
    }
    *(float4*)(el + idx * NHEAD) = elv;
    *(float4*)(er + idx * NHEAD) = erv;
}

// float atomic max via int/uint ordering trick (valid with -inf init)
__device__ __forceinline__ void atomicMaxF(float* addr, float v) {
    if (v >= 0.f) atomicMax((int*)addr, __float_as_int(v));
    else          atomicMin((unsigned int*)addr, __float_as_uint(v));
}

__device__ __forceinline__ float leaky(float v, float s) {
    return v > 0.f ? v : v * s;
}

__global__ void attn_max_kernel(const int* __restrict__ ei,
                                const float* __restrict__ el,
                                const float* __restrict__ er,
                                float* __restrict__ emax) {
    long t = (long)blockIdx.x * blockDim.x + threadIdx.x;
    if (t >= N_EDGES) return;
    int r = blockIdx.y;
    const int* src = ei + (size_t)r * 2 * N_EDGES;
    const int* dst = src + N_EDGES;
    int s = src[t], d = dst[t];
    long sb = ((long)r * N_NODES + s) * NHEAD;
    long db = ((long)r * N_NODES + d) * NHEAD;
    const float4 e4l = *(const float4*)(el + sb);
    const float4 e4r = *(const float4*)(er + db);
    const float* lp = &e4l.x;
    const float* rp = &e4r.x;
#pragma unroll
    for (int h = 0; h < NHEAD; ++h) {
        atomicMaxF(&emax[db + h], leaky(lp[h] + rp[h], SLOPE_ATTN));
    }
}

__global__ void attn_denom_kernel(const int* __restrict__ ei,
                                  const float* __restrict__ el,
                                  const float* __restrict__ er,
                                  const float* __restrict__ emax,
                                  float* __restrict__ denom) {
    long t = (long)blockIdx.x * blockDim.x + threadIdx.x;
    if (t >= N_EDGES) return;
    int r = blockIdx.y;
    const int* src = ei + (size_t)r * 2 * N_EDGES;
    const int* dst = src + N_EDGES;
    int s = src[t], d = dst[t];
    long sb = ((long)r * N_NODES + s) * NHEAD;
    long db = ((long)r * N_NODES + d) * NHEAD;
    const float4 e4l = *(const float4*)(el + sb);
    const float4 e4r = *(const float4*)(er + db);
    const float4 m4  = *(const float4*)(emax + db);
    const float* lp = &e4l.x;
    const float* rp = &e4r.x;
    const float* mp = &m4.x;
#pragma unroll
    for (int h = 0; h < NHEAD; ++h) {
        float v = leaky(lp[h] + rp[h], SLOPE_ATTN);
        float m = mp[h];
        if (!(fabsf(m) < INFINITY)) m = 0.f;   // reference: non-finite max -> 0
        atomicAdd(&denom[db + h], expf(v - m));
    }
}

// 16 threads/edge, each handles 4 consecutive floats of the 64-wide message
__global__ void attn_scatter_kernel(const int* __restrict__ ei,
                                    const float* __restrict__ el,
                                    const float* __restrict__ er,
                                    const float* __restrict__ emax,
                                    const float* __restrict__ denom,
                                    const float* __restrict__ feat,
                                    float* __restrict__ accum) {
    long t = (long)blockIdx.x * blockDim.x + threadIdx.x;
    if (t >= (long)N_EDGES * 16) return;
    int r   = blockIdx.y;
    long e  = t >> 4;
    int sub = (int)(t & 15);
    int h   = sub >> 2;
    int j   = (sub & 3) * 4;
    const int* src = ei + (size_t)r * 2 * N_EDGES;
    const int* dst = src + N_EDGES;
    int s = src[e], d = dst[e];
    long sb = (long)r * N_NODES + s;
    long db = (long)r * N_NODES + d;

    float v = leaky(el[sb * NHEAD + h] + er[db * NHEAD + h], SLOPE_ATTN);
    float m = emax[db * NHEAD + h];
    if (!(fabsf(m) < INFINITY)) m = 0.f;
    float ee    = expf(v - m);
    float alpha = ee / fmaxf(denom[db * NHEAD + h], 1e-9f);

    const float4 f = *(const float4*)(feat + sb * 64 + h * 16 + j);
    float* ap = accum + (long)d * 64 + h * 16 + j;
    atomicAdd(ap + 0, f.x * alpha);
    atomicAdd(ap + 1, f.y * alpha);
    atomicAdd(ap + 2, f.z * alpha);
    atomicAdd(ap + 3, f.w * alpha);
}

// out = ((sum_r accum) + sum_r b_r) / R, optional leaky-relu
__global__ void finalize_kernel(const float* __restrict__ accum,
                                const float* __restrict__ ball,  // Rx64
                                float* __restrict__ out, int act) {
    long i = (long)blockIdx.x * blockDim.x + threadIdx.x;
    if (i >= (long)N_NODES * 64) return;
    int j = (int)(i & 63);
    float bs = ball[j] + ball[64 + j] + ball[128 + j];
    float v = (accum[i] + bs) * (1.f / R_REL);
    if (act) v = leaky(v, SLOPE_ACT);
    out[i] = v;
}

__global__ void deg_kernel(const int* __restrict__ ei,
                           float* __restrict__ ns, float* __restrict__ nd) {
    long t = (long)blockIdx.x * blockDim.x + threadIdx.x;
    if (t >= N_EDGES) return;
    int r = blockIdx.y;
    const int* src = ei + (size_t)r * 2 * N_EDGES;
    const int* dst = src + N_EDGES;
    atomicAdd(&ns[(long)r * N_NODES + src[t]], 1.f);
    atomicAdd(&nd[(long)r * N_NODES + dst[t]], 1.f);
}

__global__ void rsqrt_kernel(float* __restrict__ p, long n) {
    long i = (long)blockIdx.x * blockDim.x + threadIdx.x;
    if (i < n) p[i] = rsqrtf(fmaxf(p[i], 1.f));
}

// GCN scatter: accum[dst] += feat_ns[r][src] * nd[r][dst]   (16 threads/edge)
__global__ void gcn_scatter_kernel(const int* __restrict__ ei,
                                   const float* __restrict__ feat,
                                   const float* __restrict__ nd,
                                   float* __restrict__ accum) {
    long t = (long)blockIdx.x * blockDim.x + threadIdx.x;
    if (t >= (long)N_EDGES * 16) return;
    int r   = blockIdx.y;
    long e  = t >> 4;
    int j   = (int)(t & 15) * 4;
    const int* src = ei + (size_t)r * 2 * N_EDGES;
    const int* dst = src + N_EDGES;
    int s = src[e], d = dst[e];
    float w = nd[(long)r * N_NODES + d];
    const float4 f = *(const float4*)(feat + ((long)r * N_NODES + s) * 64 + j);
    float* ap = accum + (long)d * 64 + j;
    atomicAdd(ap + 0, f.x * w);
    atomicAdd(ap + 1, f.y * w);
    atomicAdd(ap + 2, f.z * w);
    atomicAdd(ap + 3, f.w * w);
}

// ---------------------------------------------------------------------------
// Host orchestration
// ---------------------------------------------------------------------------
static inline dim3 g1(long n, int b) { return dim3((unsigned)((n + b - 1) / b)); }

extern "C" void kernel_launch(void* const* d_in, const int* in_sizes, int n_in,
                              void* d_out, int out_size, void* d_ws, size_t ws_size,
                              hipStream_t stream) {
    const float* x      = (const float*)d_in[0];
    const int*   ei     = (const int*)  d_in[1];
    const float* W1_gat = (const float*)d_in[2];
    const float* al1    = (const float*)d_in[3];
    const float* ar1    = (const float*)d_in[4];
    const float* b1_gat = (const float*)d_in[5];
    const float* W2_gat = (const float*)d_in[6];
    const float* al2    = (const float*)d_in[7];
    const float* ar2    = (const float*)d_in[8];
    const float* b2_gat = (const float*)d_in[9];
    const float* W1_gcn = (const float*)d_in[10];
    const float* b1_gcn = (const float*)d_in[11];
    const float* W2_gcn = (const float*)d_in[12];
    const float* b2_gcn = (const float*)d_in[13];
    const float* Wlin   = (const float*)d_in[14];
    const float* blin   = (const float*)d_in[15];
    float* out = (float*)d_out;

    // Workspace layout (floats); total ~21.9M floats (~88 MB)
    float* ws    = (float*)d_ws;
    float* feat  = ws;                                     // R*N*64
    float* accum = feat  + (size_t)R_REL * N_NODES * 64;   // N*64
    float* h_gat = accum + (size_t)N_NODES * 64;           // N*64
    float* h_gcn = h_gat + (size_t)N_NODES * 64;           // N*64
    float* el    = h_gcn + (size_t)N_NODES * 64;           // R*N*4
    float* er    = el    + (size_t)R_REL * N_NODES * NHEAD;
    float* emax  = er    + (size_t)R_REL * N_NODES * NHEAD;
    float* denom = emax  + (size_t)R_REL * N_NODES * NHEAD;
    float* ns    = denom + (size_t)R_REL * N_NODES * NHEAD; // R*N
    float* nd    = ns    + (size_t)R_REL * N_NODES;         // R*N

    const long RN   = (long)R_REL * N_NODES;
    const long RNH  = RN * NHEAD;
    const long N64  = (long)N_NODES * 64;
    const dim3 egrid(((long)N_EDGES + 255) / 256, R_REL);
    const dim3 sgrid(((long)N_EDGES * 16 + 255) / 256, R_REL);
    const dim3 ggrid(N_NODES / 16, 1, R_REL);
    const dim3 gblk(32, 4);

    auto gat_edges = [&](const float* al, const float* ar, const float* b,
                         float* hout, int act) {
        attn_dots_kernel<<<g1(RN, 256), 256, 0, stream>>>(feat, al, ar, el, er);
        fill_kernel<<<g1(RNH, 256), 256, 0, stream>>>(emax, RNH, -INFINITY);
        fill_kernel<<<g1(RNH, 256), 256, 0, stream>>>(denom, RNH, 0.f);
        fill_kernel<<<g1(N64, 256), 256, 0, stream>>>(accum, N64, 0.f);
        attn_max_kernel<<<egrid, 256, 0, stream>>>(ei, el, er, emax);
        attn_denom_kernel<<<egrid, 256, 0, stream>>>(ei, el, er, emax, denom);
        attn_scatter_kernel<<<sgrid, 256, 0, stream>>>(ei, el, er, emax, denom,
                                                       feat, accum);
        finalize_kernel<<<g1(N64, 256), 256, 0, stream>>>(accum, b, hout, act);
    };

    auto gcn_edges = [&](const float* b, float* hout, int act) {
        fill_kernel<<<g1(N64, 256), 256, 0, stream>>>(accum, N64, 0.f);
        gcn_scatter_kernel<<<sgrid, 256, 0, stream>>>(ei, feat, nd, accum);
        finalize_kernel<<<g1(N64, 256), 256, 0, stream>>>(accum, b, hout, act);
    };

    // ---- GAT branch ----
    gemm_feat_kernel<DIM_IN, false><<<ggrid, gblk, 0, stream>>>(x, W1_gat, feat, nullptr);
    gat_edges(al1, ar1, b1_gat, h_gat, 1);
    gemm_feat_kernel<HD, false><<<ggrid, gblk, 0, stream>>>(h_gat, W2_gat, feat, nullptr);
    gat_edges(al2, ar2, b2_gat, h_gat, 0);

    // ---- GCN degrees (shared by both GCN layers) ----
    fill_kernel<<<g1(RN, 256), 256, 0, stream>>>(ns, RN, 0.f);
    fill_kernel<<<g1(RN, 256), 256, 0, stream>>>(nd, RN, 0.f);
    deg_kernel<<<egrid, 256, 0, stream>>>(ei, ns, nd);
    rsqrt_kernel<<<g1(RN, 256), 256, 0, stream>>>(ns, RN);
    rsqrt_kernel<<<g1(RN, 256), 256, 0, stream>>>(nd, RN);

    // ---- GCN branch ----
    gemm_feat_kernel<DIM_IN, true><<<ggrid, gblk, 0, stream>>>(x, W1_gcn, feat, ns);
    gcn_edges(b1_gcn, h_gcn, 1);
    gemm_feat_kernel<HD, true><<<ggrid, gblk, 0, stream>>>(h_gcn, W2_gcn, feat, ns);
    gcn_edges(b2_gcn, h_gcn, 0);

    // ---- Head: [h_gat | h_gcn] @ Wlin + blin ----
    gemm_concat_kernel<<<dim3(N_NODES / 16), gblk, 0, stream>>>(h_gat, h_gcn,
                                                                Wlin, blin, out);
}
